// GPT_33397665693970
// MI455X (gfx1250) — matrix operand order repaired and checked by reading, not compiled, not run on hardware
//
#include <hip/hip_runtime.h>
#include <hip/hip_bf16.h>

// ---------------------------------------------------------------------------
// CDNA5 (gfx1250) attention block: v_wmma_f32_16x16x32_bf16 for all GEMMs,
// async global->LDS staging (ASYNCcnt) for the projection GEMMs.
// T=3072, DIM=1024, H=8, D=128, band=1408, segment boundary at T/2=1536.
// ---------------------------------------------------------------------------

typedef __bf16 bf16_t;
typedef __attribute__((ext_vector_type(16))) __bf16 v16bf;
typedef __attribute__((ext_vector_type(8)))  __bf16 v8bf;
typedef __attribute__((ext_vector_type(8)))  float  v8f;
typedef __attribute__((ext_vector_type(4)))  int    v4i;

#define TT   3072
#define DIMM 1024
#define HH   8
#define DD   128
#define BAND 1408
#define SEGB 1536

#if defined(__gfx1250__) && __has_builtin(__builtin_amdgcn_global_load_async_to_lds_b128)
#define HAVE_ASYNC_LDS 1
#else
#define HAVE_ASYNC_LDS 0
#endif

__device__ __forceinline__ v8f wmma_bf(v16bf a, v16bf b, v8f c) {
    return __builtin_amdgcn_wmma_f32_16x16x32_bf16(
        /*neg_a=*/false, a, /*neg_b=*/false, b,
        /*c_mod=*/(short)0, c, /*reuse_a=*/false, /*reuse_b=*/false);
}

// A-matrix 16x32 bf16 fragment (ISA 7.12.2): lane m = L%16,
// element e -> k = (e/8)*16 + (L/16)*8 + e%8
__device__ __forceinline__ v16bf load_frag_a(const bf16_t* base, int ld) {
    int lane = threadIdx.x & 31;
    int m = lane & 15, half = lane >> 4;
    const bf16_t* p = base + (size_t)m * ld + half * 8;
    v8bf lo = *(const v8bf*)(p);
    v8bf hi = *(const v8bf*)(p + 16);
    v16bf r;
#pragma unroll
    for (int i = 0; i < 8; ++i) { r[i] = lo[i]; r[i + 8] = hi[i]; }
    return r;
}

// B-matrix 32x16 bf16 fragment: lane n = L%16, element e -> k = (e/8)*16 + (L/16)*8 + e%8.
// Source is stored [N][K] row-major (i.e. column n of B contiguous in K).
__device__ __forceinline__ v16bf load_frag_b(const bf16_t* base, int ld) {
    int lane = threadIdx.x & 31;
    int n = lane & 15, half = lane >> 4;
    const bf16_t* p = base + (size_t)n * ld + half * 8;
    v8bf lo = *(const v8bf*)(p);
    v8bf hi = *(const v8bf*)(p + 16);
    v16bf r;
#pragma unroll
    for (int i = 0; i < 8; ++i) { r[i] = lo[i]; r[i + 8] = hi[i]; }
    return r;
}

__device__ __forceinline__ float sigf(float z) { return 1.f / (1.f + __expf(-z)); }

#if HAVE_ASYNC_LDS
__device__ __forceinline__ void async_cp16(const void* g, void* l) {
    __builtin_amdgcn_global_load_async_to_lds_b128(
        (__attribute__((address_space(1))) v4i*)g,
        (__attribute__((address_space(3))) v4i*)l,
        /*offset=*/0, /*cpol=*/0);
}
#endif

// ---------------------------------------------------------------------------
// f32 -> bf16 (optionally scaled by a device scalar)
// ---------------------------------------------------------------------------
__global__ void k_cvt_bf16(const float* __restrict__ src, bf16_t* __restrict__ dst,
                           const float* scale_ptr, int n) {
    int i = blockIdx.x * blockDim.x + threadIdx.x;
    if (i >= n) return;
    float s = scale_ptr ? scale_ptr[0] : 1.f;
    dst[i] = (bf16_t)(src[i] * s);
}

// ---------------------------------------------------------------------------
// C[M][N] = A[M][K] * B[N][K]^T   (bf16 in, f32 out)
// 64x64 block tile, 4 waves (each 32x32). Double-buffered async LDS staging:
// per K-step each thread issues 4 global_load_async_to_lds_b128 (8KB tiles),
// s_wait_asynccnt 4 keeps one buffer in flight while computing the other.
// ---------------------------------------------------------------------------
__global__ __launch_bounds__(128) void k_gemm_xwT(const bf16_t* __restrict__ A,
                                                  const bf16_t* __restrict__ B,
                                                  float* __restrict__ C,
                                                  int M, int N, int K) {
#if HAVE_ASYNC_LDS
    __shared__ bf16_t sA[2][64 * 32];
    __shared__ bf16_t sB[2][64 * 32];
    int tid = threadIdx.x;
    int w = tid >> 5, lane = tid & 31;
    int m_blk = blockIdx.y * 64;
    int n_blk = blockIdx.x * 64;
    int m0 = (w >> 1) * 32;
    int n0 = (w & 1) * 32;
    int srow  = tid >> 2;          // 0..31
    int sbyte = (tid & 3) * 16;    // 0/16/32/48 within a 64B row

    auto issue = [&](int buf, int k0) {
        const char* ga0 = (const char*)(A + (size_t)(m_blk + srow) * K + k0) + sbyte;
        const char* ga1 = (const char*)(A + (size_t)(m_blk + 32 + srow) * K + k0) + sbyte;
        const char* gb0 = (const char*)(B + (size_t)(n_blk + srow) * K + k0) + sbyte;
        const char* gb1 = (const char*)(B + (size_t)(n_blk + 32 + srow) * K + k0) + sbyte;
        char* la0 = (char*)&sA[buf][(size_t)srow * 32] + sbyte;
        char* la1 = (char*)&sA[buf][(size_t)(32 + srow) * 32] + sbyte;
        char* lb0 = (char*)&sB[buf][(size_t)srow * 32] + sbyte;
        char* lb1 = (char*)&sB[buf][(size_t)(32 + srow) * 32] + sbyte;
        async_cp16(ga0, la0);
        async_cp16(ga1, la1);
        async_cp16(gb0, lb0);
        async_cp16(gb1, lb1);
    };

    v8f acc00 = {}, acc01 = {}, acc10 = {}, acc11 = {};
    int nk = K >> 5;
    issue(0, 0);
    for (int i = 0; i < nk; ++i) {
        int cur = i & 1;
        if (i + 1 < nk) {
            issue(cur ^ 1, (i + 1) << 5);
            // in-order async completion: <=4 outstanding means buffer `cur` landed
            asm volatile("s_wait_asynccnt 4" ::: "memory");
        } else {
            asm volatile("s_wait_asynccnt 0" ::: "memory");
        }
        __syncthreads();
        v16bf a0 = load_frag_a(&sA[cur][(size_t)m0 * 32], 32);
        v16bf a1 = load_frag_a(&sA[cur][(size_t)(m0 + 16) * 32], 32);
        v16bf b0 = load_frag_b(&sB[cur][(size_t)n0 * 32], 32);
        v16bf b1 = load_frag_b(&sB[cur][(size_t)(n0 + 16) * 32], 32);
        acc00 = wmma_bf(a0, b0, acc00);
        acc01 = wmma_bf(a0, b1, acc01);
        acc10 = wmma_bf(a1, b0, acc10);
        acc11 = wmma_bf(a1, b1, acc11);
        __syncthreads();   // reads done before next issue overwrites this buffer
    }
#else
    int w = threadIdx.x >> 5;
    int lane = threadIdx.x & 31;
    int m0 = blockIdx.y * 64 + (w >> 1) * 32;
    int n0 = blockIdx.x * 64 + (w & 1) * 32;
    v8f acc00 = {}, acc01 = {}, acc10 = {}, acc11 = {};
    for (int k0 = 0; k0 < K; k0 += 32) {
        v16bf a0 = load_frag_a(A + (size_t)m0 * K + k0, K);
        v16bf a1 = load_frag_a(A + (size_t)(m0 + 16) * K + k0, K);
        v16bf b0 = load_frag_b(B + (size_t)n0 * K + k0, K);
        v16bf b1 = load_frag_b(B + (size_t)(n0 + 16) * K + k0, K);
        acc00 = wmma_bf(a0, b0, acc00);
        acc01 = wmma_bf(a0, b1, acc01);
        acc10 = wmma_bf(a1, b0, acc10);
        acc11 = wmma_bf(a1, b1, acc11);
    }
#endif
    int col = lane & 15, half = lane >> 4;
#if HAVE_ASYNC_LDS
    int mo = m_blk + m0, no = n_blk + n0;
#else
    int mo = m0, no = n0;
#endif
#pragma unroll
    for (int r = 0; r < 8; ++r) {
        int rr = r + 8 * half;
        C[(size_t)(mo + rr) * N + no + col]           = acc00[r];
        C[(size_t)(mo + rr) * N + no + 16 + col]      = acc01[r];
        C[(size_t)(mo + 16 + rr) * N + no + col]      = acc10[r];
        C[(size_t)(mo + 16 + rr) * N + no + 16 + col] = acc11[r];
    }
}

// ---------------------------------------------------------------------------
// Postprocess qkv: rms-norm + rotary (q,k), k channel-shift with k(t-1),
// ve-gated v, attention gate. One wave per (t, h); lane owns dims
// d = lane, lane+32, lane+64, lane+96.
// Outputs: qb[h][t][d] (x0.1), kb[h][t][d], vT[h][d][t], gate[t][h].
// ---------------------------------------------------------------------------
__device__ __forceinline__ float wave_sumsq(float a, float b, float c, float d) {
    float s = a * a + b * b + c * c + d * d;
    s += __shfl_xor(s, 16, 32);
    s += __shfl_xor(s, 8, 32);
    s += __shfl_xor(s, 4, 32);
    s += __shfl_xor(s, 2, 32);
    s += __shfl_xor(s, 1, 32);
    return s;
}

__device__ __forceinline__ void norm_rot_row(const float* __restrict__ row,
                                             const float* __restrict__ cosb,
                                             const float* __restrict__ sinb,
                                             int t, int lane, float out[4]) {
    float a = row[lane], b = row[lane + 32], c = row[lane + 64], d = row[lane + 96];
    float ss = wave_sumsq(a, b, c, d);
    float sc = rsqrtf(ss * (1.f / (float)DD) + 1e-6f);
    a *= sc; b *= sc; c *= sc; d *= sc;
    float c0 = cosb[t * 64 + lane],      s0 = sinb[t * 64 + lane];
    float c1 = cosb[t * 64 + lane + 32], s1 = sinb[t * 64 + lane + 32];
    out[0] =  a * c0 + c * s0;   // d = lane
    out[2] = -a * s0 + c * c0;   // d = lane+64
    out[1] =  b * c1 + d * s1;   // d = lane+32
    out[3] = -b * s1 + d * c1;   // d = lane+96
}

__global__ __launch_bounds__(128) void k_postproc(const float* __restrict__ qkv,
                                                  const float* __restrict__ x,
                                                  const float* __restrict__ ve,
                                                  const float* __restrict__ attn_gw,
                                                  const float* __restrict__ ve_gw,
                                                  const float* __restrict__ cosb,
                                                  const float* __restrict__ sinb,
                                                  bf16_t* __restrict__ qb,
                                                  bf16_t* __restrict__ kb,
                                                  bf16_t* __restrict__ vT,
                                                  float* __restrict__ gate) {
    int gid = blockIdx.x * (blockDim.x >> 5) + (threadIdx.x >> 5);
    int lane = threadIdx.x & 31;
    int t = gid >> 3, h = gid & 7;
    if (t >= TT) return;

    // ---- q ----
    float q[4];
    norm_rot_row(qkv + (size_t)t * (3 * DIMM) + h * DD, cosb, sinb, t, lane, q);
    bf16_t* qd = qb + ((size_t)h * TT + t) * DD;
    qd[lane]      = (bf16_t)(0.1f * q[0]);
    qd[lane + 32] = (bf16_t)(0.1f * q[1]);
    qd[lane + 64] = (bf16_t)(0.1f * q[2]);
    qd[lane + 96] = (bf16_t)(0.1f * q[3]);

    // ---- k (with channel shift from k(t-1)) ----
    float kt[4], kp[4];
    norm_rot_row(qkv + (size_t)t * (3 * DIMM) + (HH + h) * DD, cosb, sinb, t, lane, kt);
    int tp = (t > 0) ? (t - 1) : 0;
    norm_rot_row(qkv + (size_t)tp * (3 * DIMM) + (HH + h) * DD, cosb, sinb, tp, lane, kp);
    bf16_t* kd = kb + ((size_t)h * TT + t) * DD;
    kd[lane]      = (bf16_t)kt[0];   // d in [0,32)   : k(t)
    kd[lane + 32] = (bf16_t)kp[1];   // d in [32,64)  : k(t-1)
    kd[lane + 64] = (bf16_t)kt[2];   // d in [64,96)  : k(t)
    kd[lane + 96] = (bf16_t)kp[3];   // d in [96,128) : k(t-1)

    // ---- gates ----
    float gv = 0.f, ga = 0.f;
#pragma unroll
    for (int j = 0; j < 16; ++j) {
        float xv = x[(size_t)t * DIMM + j];
        gv += xv * ve_gw[h * 16 + j];
        ga += xv * attn_gw[h * 16 + j];
    }
    float vg = 2.f * sigf(gv);
    if (lane == 0) gate[t * HH + h] = sigf(ga);

    // ---- v (transposed store: vT[h][d][t]) ----
    const float* vr  = qkv + (size_t)t * (3 * DIMM) + (2 * HH + h) * DD;
    const float* ver = ve + (size_t)t * DIMM + h * DD;
#pragma unroll
    for (int i = 0; i < 4; ++i) {
        int d = lane + 32 * i;
        vT[((size_t)h * DD + d) * TT + t] = (bf16_t)(vr[d] + vg * ver[d]);
    }
}

// ---------------------------------------------------------------------------
// Banded causal attention. One wave per 16-row query tile (4 waves/block).
// Scores via WMMA (q already carries ATTN_SCALE). No max-subtraction:
// scores are bounded by |q||k|*0.1 <= 12.8, exp fits f32/bf16 easily.
// P goes C-layout -> LDS -> A-layout bf16 for the PV WMMAs against vT.
// ---------------------------------------------------------------------------
__global__ __launch_bounds__(128) void k_attn(const bf16_t* __restrict__ qb,
                                              const bf16_t* __restrict__ kb,
                                              const bf16_t* __restrict__ vT,
                                              const float* __restrict__ gate,
                                              bf16_t* __restrict__ yb) {
    __shared__ float pl[4][16 * 32];
    int w = threadIdx.x >> 5, lane = threadIdx.x & 31;
    int h = blockIdx.y;
    int qt = blockIdx.x * 4 + w;
    int t0 = qt * 16;
    int seg_lo = (t0 >= SEGB) ? SEGB : 0;
    int s_start = t0 - BAND; if (s_start < seg_lo) s_start = seg_lo;
    int s_end = t0 + 16;
    int col = lane & 15, half = lane >> 4;
    float* myp = pl[w];

    v16bf qf[4];
    const bf16_t* qbase = qb + ((size_t)h * TT + t0) * DD;
#pragma unroll
    for (int f = 0; f < 4; ++f) qf[f] = load_frag_a(qbase + f * 32, DD);

    v8f yacc[8] = {};
    float lsum[8] = {0.f, 0.f, 0.f, 0.f, 0.f, 0.f, 0.f, 0.f};

    for (int s0 = s_start; s0 < s_end; s0 += 32) {
        bool two = (s0 + 16) < s_end;
        v8f sa0 = {}, sa1 = {};
#pragma unroll
        for (int f = 0; f < 4; ++f) {
            v16bf b0 = load_frag_b(kb + ((size_t)h * TT + s0) * DD + f * 32, DD);
            sa0 = wmma_bf(qf[f], b0, sa0);
        }
        if (two) {
#pragma unroll
            for (int f = 0; f < 4; ++f) {
                v16bf b1 = load_frag_b(kb + ((size_t)h * TT + s0 + 16) * DD + f * 32, DD);
                sa1 = wmma_bf(qf[f], b1, sa1);
            }
        }
        // mask + exp, stage P tile (16x32) in LDS (C layout -> array)
#pragma unroll
        for (int r = 0; r < 8; ++r) {
            int t = t0 + r + 8 * half;
            int s = s0 + col;
            float e0 = (s <= t && s >= t - BAND) ? __expf(sa0[r]) : 0.f;
            int s1 = s + 16;
            float e1 = (two && s1 <= t && s1 >= t - BAND) ? __expf(sa1[r]) : 0.f;
            lsum[r] += e0 + e1;
            myp[(r + 8 * half) * 32 + col] = e0;
            myp[(r + 8 * half) * 32 + 16 + col] = e1;
        }
        __asm__ volatile("" ::: "memory"); // keep LDS writes before reads (HW DS in-order per wave)
        // re-read as A fragment (16x32 bf16)
        v16bf pf;
        {
            const float* pr = myp + (lane & 15) * 32 + half * 8;
#pragma unroll
            for (int i = 0; i < 8; ++i) {
                pf[i]     = (bf16_t)pr[i];
                pf[i + 8] = (bf16_t)pr[16 + i];
            }
        }
        // y += P * V  (B-frags from vT[h][d][t], contiguous in s)
#pragma unroll
        for (int nt = 0; nt < 8; ++nt) {
            v16bf vb = load_frag_b(vT + ((size_t)h * DD + nt * 16) * TT + s0, TT);
            yacc[nt] = wmma_bf(pf, vb, yacc[nt]);
        }
        __asm__ volatile("" ::: "memory");
    }

    // row sums: reduce across the 16 lanes sharing each half
#pragma unroll
    for (int r = 0; r < 8; ++r) {
        lsum[r] += __shfl_xor(lsum[r], 1, 32);
        lsum[r] += __shfl_xor(lsum[r], 2, 32);
        lsum[r] += __shfl_xor(lsum[r], 4, 32);
        lsum[r] += __shfl_xor(lsum[r], 8, 32);
    }

#pragma unroll
    for (int r = 0; r < 8; ++r) {
        int t = t0 + r + 8 * half;
        float inv = gate[t * HH + h] / lsum[r];
#pragma unroll
        for (int nt = 0; nt < 8; ++nt) {
            int d = nt * 16 + col;
            yb[(size_t)t * DIMM + h * DD + d] = (bf16_t)(yacc[nt][r] * inv);
        }
    }
}

// ---------------------------------------------------------------------------
// Host-side launcher
// ---------------------------------------------------------------------------
extern "C" void kernel_launch(void* const* d_in, const int* in_sizes, int n_in,
                              void* d_out, int out_size, void* d_ws, size_t ws_size,
                              hipStream_t stream) {
    const float* x        = (const float*)d_in[0];
    const float* qkvo_w   = (const float*)d_in[1];
    const float* lambdas  = (const float*)d_in[2];
    const float* ve       = (const float*)d_in[3];
    const float* attn_gw  = (const float*)d_in[4];
    const float* ve_gw    = (const float*)d_in[5];
    const float* cosb     = (const float*)d_in[6];
    const float* sinb     = (const float*)d_in[7];
    float* out            = (float*)d_out;
    (void)in_sizes; (void)n_in; (void)out_size; (void)ws_size;

    char* ws = (char*)d_ws;
    size_t off = 0;
    auto alloc = [&](size_t bytes) {
        size_t p = off;
        off = (off + bytes + 255) & ~(size_t)255;
        return p;
    };
    bf16_t* xb    = (bf16_t*)(ws + alloc((size_t)TT * DIMM * 2));
    bf16_t* wqkv  = (bf16_t*)(ws + alloc((size_t)3 * DIMM * DIMM * 2));
    bf16_t* wo    = (bf16_t*)(ws + alloc((size_t)DIMM * DIMM * 2));
    float*  qkv   = (float*) (ws + alloc((size_t)TT * 3 * DIMM * 4));
    bf16_t* qb    = (bf16_t*)(ws + alloc((size_t)HH * TT * DD * 2));
    bf16_t* kb    = (bf16_t*)(ws + alloc((size_t)HH * TT * DD * 2));
    bf16_t* vT    = (bf16_t*)(ws + alloc((size_t)HH * DD * TT * 2 + 256)); // pad: last pair overreads <=32B
    float*  gate  = (float*) (ws + alloc((size_t)TT * HH * 4));
    bf16_t* yb    = (bf16_t*)(ws + alloc((size_t)TT * DIMM * 2));

    // 1) bf16 conversions
    {
        int n = TT * DIMM;
        k_cvt_bf16<<<(n + 255) / 256, 256, 0, stream>>>(x, xb, nullptr, n);
        n = 3 * DIMM * DIMM;
        k_cvt_bf16<<<(n + 255) / 256, 256, 0, stream>>>(qkvo_w, wqkv, lambdas + 0, n);
        n = DIMM * DIMM;
        k_cvt_bf16<<<(n + 255) / 256, 256, 0, stream>>>(qkvo_w + (size_t)3 * DIMM * DIMM, wo,
                                                        lambdas + 1, n);
    }
    // 2) qkv = x @ (l0*Wqkv)^T   [3072 x 3072]
    k_gemm_xwT<<<dim3(3 * DIMM / 64, TT / 64), 128, 0, stream>>>(xb, wqkv, qkv,
                                                                 TT, 3 * DIMM, DIMM);
    // 3) postprocess -> qb, kb, vT, gate
    k_postproc<<<(TT * HH) / 4, 128, 0, stream>>>(qkv, x, ve, attn_gw, ve_gw,
                                                  cosb, sinb, qb, kb, vT, gate);
    // 4) banded attention -> yb (gated)
    k_attn<<<dim3(TT / 64, HH), 128, 0, stream>>>(qb, kb, vT, gate, yb);
    // 5) out = y @ (l1*Wo)^T   [3072 x 1024]
    k_gemm_xwT<<<dim3(DIMM / 64, TT / 64), 128, 0, stream>>>(yb, wo, out,
                                                             TT, DIMM, DIMM);
}
